// DiSAN_84155589198268
// MI455X (gfx1250) — matrix operand order repaired and checked by reading, not compiled, not run on hardware
//
#include <hip/hip_runtime.h>
#include <hip/hip_bf16.h>

typedef __attribute__((ext_vector_type(16))) __bf16 v16bf;
typedef __attribute__((ext_vector_type(8)))  __bf16 v8bf;
typedef __attribute__((ext_vector_type(8)))  float  v8f;

#define DIMN 200
#define BB   8
#define LL   256
#define DP   224          // DIM padded to multiple of 32
#define D2   400
#define D2P  448          // 2*DIM padded to multiple of 32
#define ROWS (BB*LL)      // 2048
#define NEG_INF (-1.0e13f)

// ---------------- pack: W[K,N] f32 -> W^T padded bf16 [Np,Kp] ----------------
__global__ void k_pack_w(const float* __restrict__ W, __bf16* __restrict__ BT,
                         int K, int N, int Kp, int Np) {
    int i = blockIdx.x * blockDim.x + threadIdx.x;
    int total = Np * Kp;
    if (i >= total) return;
    int n = i / Kp, k = i % Kp;
    float v = (n < N && k < K) ? W[(size_t)k * N + n] : 0.0f;
    BT[i] = (__bf16)v;
}

__global__ void k_pack_b(const float* __restrict__ src, float* __restrict__ dst,
                         int N, int Np) {
    int i = blockIdx.x * blockDim.x + threadIdx.x;
    if (i < Np) dst[i] = (i < N) ? src[i] : 0.0f;
}

// ---------------- embedding gather + pad mask ----------------
__global__ void k_embed(const int* __restrict__ x, const float* __restrict__ emb,
                        __bf16* __restrict__ xxbf, unsigned char* __restrict__ mask) {
    int bl = blockIdx.x;
    int e  = threadIdx.x;
    if (e >= DP) return;
    int idx = x[bl];
    float v = (e < DIMN) ? emb[(size_t)idx * DIMN + e] : 0.0f;
    xxbf[(size_t)bl * DP + e] = (__bf16)v;
    mask[(size_t)bl * DP + e] = (e < DIMN && v == 1.0f) ? 1 : 0;
}

// ---------------- WMMA GEMM: D = act(A[M,K] * BT[N,K]^T (+C) (+bias)) ----------------
// one wave32 per 16x16 tile; KT templated so the K-loop fully unrolls into
// back-to-back v_wmma_f32_16x16x32_bf16.
template <int KT>
__global__ void k_gemm(const __bf16* __restrict__ A, const __bf16* __restrict__ BT,
                       const float* __restrict__ Cin, const float* __restrict__ bias,
                       float* __restrict__ D, __bf16* __restrict__ Dbf,
                       int N, int act) {
    int m0   = blockIdx.x * 16;
    int n0   = blockIdx.y * 16;
    int lane = threadIdx.x;          // 0..31, wave32
    int r16  = lane & 15;
    int half = lane >> 4;

    const __bf16* ap = A  + (size_t)(m0 + r16) * KT;   // A row for this lane
    const __bf16* bp = BT + (size_t)(n0 + r16) * KT;   // B column (row of BT)

    int mrowBase = m0 + (half << 3);
    int ncol     = n0 + r16;

    v8f acc = {0.f,0.f,0.f,0.f,0.f,0.f,0.f,0.f};
    if (Cin) {
#pragma unroll
        for (int r = 0; r < 8; ++r)
            acc[r] = Cin[(size_t)(mrowBase + r) * N + ncol];
    }

#pragma unroll
    for (int k = 0; k < KT; k += 32) {
        // A 16x32 bf16 fragment: lane half selects 8-wide K chunks (ISA layout)
        const __bf16* apk = ap + k + (half << 3);
        v8bf a0 = *(const v8bf*)(apk);
        v8bf a1 = *(const v8bf*)(apk + 16);
        v16bf av;
#pragma unroll
        for (int i = 0; i < 8; ++i) { av[i] = a0[i]; av[i + 8] = a1[i]; }
        // B 32x16 bf16 fragment: lane half selects 8-wide K chunks (ISA layout)
        v16bf bv = __builtin_shufflevector(*(const v8bf*)(bp + k + (half << 3)), *(const v8bf*)(bp + k + (half << 3) + 16), 0,1,2,3,4,5,6,7,8,9,10,11,12,13,14,15);
        acc = __builtin_amdgcn_wmma_f32_16x16x32_bf16(
                  false, av, false, bv, (short)0, acc, false, false);
    }

    float bvadd = bias ? bias[ncol] : 0.0f;
#pragma unroll
    for (int r = 0; r < 8; ++r) {
        float v = acc[r] + bvadd;
        if (act == 1) v = (v > 0.0f) ? v : (__expf(v) - 1.0f);   // ELU
        size_t o = (size_t)(mrowBase + r) * N + ncol;
        D[o] = v;
        if (Dbf) Dbf[o] = (__bf16)v;
    }
}

// ---------------- fused per-channel attention (fw+bw), online softmax ----------------
__global__ void k_attn(const float* __restrict__ h1, const float* __restrict__ h2,
                       const float* __restrict__ h,  const unsigned char* __restrict__ mask,
                       const float* __restrict__ bp, const float* __restrict__ cptr,
                       float* __restrict__ sfw, __bf16* __restrict__ sfwb,
                       float* __restrict__ sbw, __bf16* __restrict__ sbwb) {
    int bl = blockIdx.x;
    int b  = bl >> 8;
    int l  = bl & 255;
    int e  = threadIdx.x;
    if (e >= DP) return;

    size_t base = (size_t)(b * LL) * DP + e;
    float cval  = cptr[0];
    float invc  = 1.0f / cval;
    float pre   = h1[(size_t)bl * DP + e] + bp[e];
    unsigned char mym = mask[(size_t)bl * DP + e];

    float Mf = -3.0e38f, Sf = 0.f, Af = 0.f;
    float Mb = -3.0e38f, Sb = 0.f, Ab = 0.f;

    for (int m = 0; m < LL; ++m) {
        size_t off = base + (size_t)m * DP;
        if (m + 8 < LL) {                       // global_prefetch_b8 path
            __builtin_prefetch(h2 + off + 8 * DP, 0, 1);
            __builtin_prefetch(h  + off + 8 * DP, 0, 1);
        }
        float h2v = h2[off];
        float hv  = h[off];
        int   mk  = mym | mask[off];
        float t = cval * tanhf((pre + h2v) * invc);
        if (mk) t = NEG_INF;
        float tf = (m <= l) ? NEG_INF : t;      // forward keeps m > l
        float tb = (m >= l) ? NEG_INF : t;      // backward keeps m < l

        float nM = fmaxf(Mf, tf);
        float sc = __expf(Mf - nM);
        float p  = __expf(tf - nM);
        Sf = Sf * sc + p;  Af = Af * sc + p * hv;  Mf = nM;

        nM = fmaxf(Mb, tb);
        sc = __expf(Mb - nM);
        p  = __expf(tb - nM);
        Sb = Sb * sc + p;  Ab = Ab * sc + p * hv;  Mb = nM;
    }

    float vfw = Af / Sf;
    float vbw = Ab / Sb;
    size_t o = (size_t)bl * DP + e;
    sfw[o] = vfw;  sfwb[o] = (__bf16)vfw;
    sbw[o] = vbw;  sbwb[o] = (__bf16)vbw;
}

// ---------------- sigmoid gate + fusion, build uu [2048, 448] ----------------
__global__ void k_combine(const float* __restrict__ Gf, const float* __restrict__ Gb,
                          const float* __restrict__ h,
                          const float* __restrict__ sfw, const float* __restrict__ sbw,
                          float* __restrict__ uu, __bf16* __restrict__ uubf) {
    int bl = blockIdx.x;
    int e  = threadIdx.x;
    size_t rb = (size_t)bl * D2P;
    if (e < DIMN) {
        size_t o = (size_t)bl * DP + e;
        float hv  = h[o];
        float ffw = 1.0f / (1.0f + __expf(-Gf[o]));
        float fbw = 1.0f / (1.0f + __expf(-Gb[o]));
        float ufw = ffw * hv + (1.0f - ffw) * sfw[o];
        float ubw = fbw * hv + (1.0f - fbw) * sbw[o];
        uu[rb + e]          = ufw;  uubf[rb + e]          = (__bf16)ufw;
        uu[rb + DIMN + e]   = ubw;  uubf[rb + DIMN + e]   = (__bf16)ubw;
    } else if (e < DIMN + 48) {     // zero pad cols 400..447
        uu[rb + DIMN + e]   = 0.0f; uubf[rb + DIMN + e]   = (__bf16)0.0f;
    }
}

// ---------------- s_s[b,e] = sum_l uu*att_s ----------------
__global__ void k_ss(const float* __restrict__ uu, const float* __restrict__ atts,
                     float* __restrict__ ss) {
    int b = blockIdx.x;
    int e = threadIdx.x;
    if (e >= D2P) return;
    size_t base = (size_t)b * LL * D2P + e;
    float acc = 0.f;
    for (int l = 0; l < LL; ++l)
        acc += uu[base + (size_t)l * D2P] * atts[base + (size_t)l * D2P];
    ss[b * D2P + e] = acc;
}

// ---------------- head: y = relu(ss@F1+b)@F2+b ----------------
__global__ void k_final(const float* __restrict__ ss, const float* __restrict__ F1w,
                        const float* __restrict__ F1b, const float* __restrict__ F2w,
                        const float* __restrict__ F2b, float* __restrict__ y) {
    __shared__ float red[256];
    int b = blockIdx.x;
    int j = threadIdx.x;
    float r = 0.f;
    if (j < DIMN) {
        float acc = F1b[j];
        const float* s = ss + b * D2P;
        for (int k = 0; k < D2; ++k) acc += s[k] * F1w[(size_t)k * DIMN + j];
        r = fmaxf(acc, 0.f) * F2w[j];
    }
    red[j] = r;
    __syncthreads();
    for (int s2 = 128; s2 > 0; s2 >>= 1) {
        if (j < s2) red[j] += red[j + s2];
        __syncthreads();
    }
    if (j == 0) y[b] = red[0] + F2b[0];
}

// ======================================================================
extern "C" void kernel_launch(void* const* d_in, const int* in_sizes, int n_in,
                              void* d_out, int out_size, void* d_ws, size_t ws_size,
                              hipStream_t stream) {
    (void)in_sizes; (void)n_in; (void)out_size; (void)ws_size;
    const int*   x    = (const int*)  d_in[0];
    const float* emb  = (const float*)d_in[1];
    const float* Whw  = (const float*)d_in[2];
    const float* Whb  = (const float*)d_in[3];
    const float* W1w  = (const float*)d_in[4];
    const float* W2w  = (const float*)d_in[5];
    const float* bvec = (const float*)d_in[6];
    const float* cp   = (const float*)d_in[7];
    const float* Wf1w = (const float*)d_in[8];
    const float* Wf2w = (const float*)d_in[9];
    const float* Wf2b = (const float*)d_in[10];
    const float* Ws1w = (const float*)d_in[11];
    const float* Ws1b = (const float*)d_in[12];
    const float* Wsw  = (const float*)d_in[13];
    const float* Wsb  = (const float*)d_in[14];
    const float* F1w  = (const float*)d_in[15];
    const float* F1b  = (const float*)d_in[16];
    const float* F2w  = (const float*)d_in[17];
    const float* F2b  = (const float*)d_in[18];
    float* y = (float*)d_out;

    char* w = (char*)d_ws;
    size_t off = 0;
    auto alloc = [&](size_t bytes) -> char* {
        char* p = w + off;
        off = (off + bytes + 255) & ~(size_t)255;
        return p;
    };

    __bf16*        xxbf  = (__bf16*)       alloc((size_t)ROWS * DP * 2);
    unsigned char* maskB = (unsigned char*)alloc((size_t)ROWS * DP);
    float*  hF   = (float*) alloc((size_t)ROWS * DP * 4);
    __bf16* hBf  = (__bf16*)alloc((size_t)ROWS * DP * 2);
    float*  h1F  = (float*) alloc((size_t)ROWS * DP * 4);
    float*  h2F  = (float*) alloc((size_t)ROWS * DP * 4);
    float*  sfw  = (float*) alloc((size_t)ROWS * DP * 4);
    __bf16* sfwB = (__bf16*)alloc((size_t)ROWS * DP * 2);
    float*  sbw  = (float*) alloc((size_t)ROWS * DP * 4);
    __bf16* sbwB = (__bf16*)alloc((size_t)ROWS * DP * 2);
    float*  Gf   = (float*) alloc((size_t)ROWS * DP * 4);
    float*  Gb   = (float*) alloc((size_t)ROWS * DP * 4);
    float*  uu   = (float*) alloc((size_t)ROWS * D2P * 4);
    __bf16* uuB  = (__bf16*)alloc((size_t)ROWS * D2P * 2);
    float*  tF   = (float*) alloc((size_t)ROWS * D2P * 4);
    __bf16* tBf  = (__bf16*)alloc((size_t)ROWS * D2P * 2);
    float*  atts = (float*) alloc((size_t)ROWS * D2P * 4);
    float*  ssb  = (float*) alloc((size_t)BB * D2P * 4);
    __bf16* WhT  = (__bf16*)alloc((size_t)DP * DP * 2);
    __bf16* W1T  = (__bf16*)alloc((size_t)DP * DP * 2);
    __bf16* W2T  = (__bf16*)alloc((size_t)DP * DP * 2);
    __bf16* Wf1T = (__bf16*)alloc((size_t)DP * DP * 2);
    __bf16* Wf2T = (__bf16*)alloc((size_t)DP * DP * 2);
    __bf16* Ws1T = (__bf16*)alloc((size_t)D2P * D2P * 2);
    __bf16* WsT  = (__bf16*)alloc((size_t)D2P * D2P * 2);
    float* WhbP  = (float*)alloc(DP * 4);
    float* bP    = (float*)alloc(DP * 4);
    float* Wf2bP = (float*)alloc(DP * 4);
    float* Ws1bP = (float*)alloc(D2P * 4);
    float* WsbP  = (float*)alloc(D2P * 4);

    // ---- pack weights (transpose + pad + bf16) and biases ----
    int nSmall = (DP * DP + 255) / 256;
    int nBig   = (D2P * D2P + 255) / 256;
    k_pack_w<<<nSmall, 256, 0, stream>>>(Whw,  WhT,  DIMN, DIMN, DP,  DP);
    k_pack_w<<<nSmall, 256, 0, stream>>>(W1w,  W1T,  DIMN, DIMN, DP,  DP);
    k_pack_w<<<nSmall, 256, 0, stream>>>(W2w,  W2T,  DIMN, DIMN, DP,  DP);
    k_pack_w<<<nSmall, 256, 0, stream>>>(Wf1w, Wf1T, DIMN, DIMN, DP,  DP);
    k_pack_w<<<nSmall, 256, 0, stream>>>(Wf2w, Wf2T, DIMN, DIMN, DP,  DP);
    k_pack_w<<<nBig,   256, 0, stream>>>(Ws1w, Ws1T, D2,   D2,   D2P, D2P);
    k_pack_w<<<nBig,   256, 0, stream>>>(Wsw,  WsT,  D2,   D2,   D2P, D2P);
    k_pack_b<<<1, 256, 0, stream>>>(Whb,  WhbP,  DIMN, DP);
    k_pack_b<<<1, 256, 0, stream>>>(bvec, bP,    DIMN, DP);
    k_pack_b<<<1, 256, 0, stream>>>(Wf2b, Wf2bP, DIMN, DP);
    k_pack_b<<<2, 256, 0, stream>>>(Ws1b, Ws1bP, D2,   D2P);
    k_pack_b<<<2, 256, 0, stream>>>(Wsb,  WsbP,  D2,   D2P);

    // ---- embedding + mask ----
    k_embed<<<ROWS, 256, 0, stream>>>(x, emb, xxbf, maskB);

    dim3 gS(ROWS / 16, DP / 16);    // 128 x 14
    dim3 gB(ROWS / 16, D2P / 16);   // 128 x 28
    dim3 wv(32);

    // h = elu(xx @ Wh + b)  (f32 + bf16 copies)
    k_gemm<DP><<<gS, wv, 0, stream>>>(xxbf, WhT, nullptr, WhbP, hF, hBf, DP, 1);
    // h1 = h @ W1 ; h2 = h @ W2
    k_gemm<DP><<<gS, wv, 0, stream>>>(hBf, W1T, nullptr, nullptr, h1F, nullptr, DP, 0);
    k_gemm<DP><<<gS, wv, 0, stream>>>(hBf, W2T, nullptr, nullptr, h2F, nullptr, DP, 0);

    // fused per-channel attention, both directions
    k_attn<<<ROWS, 256, 0, stream>>>(h1F, h2F, hF, maskB, bP, cp,
                                     sfw, sfwB, sbw, sbwB);

    // G = s_ @ Wf1 ; G += h @ Wf2 + b  (accumulate-into-C GEMM)
    k_gemm<DP><<<gS, wv, 0, stream>>>(sfwB, Wf1T, nullptr, nullptr, Gf, nullptr, DP, 0);
    k_gemm<DP><<<gS, wv, 0, stream>>>(hBf,  Wf2T, Gf,      Wf2bP,  Gf, nullptr, DP, 0);
    k_gemm<DP><<<gS, wv, 0, stream>>>(sbwB, Wf1T, nullptr, nullptr, Gb, nullptr, DP, 0);
    k_gemm<DP><<<gS, wv, 0, stream>>>(hBf,  Wf2T, Gb,      Wf2bP,  Gb, nullptr, DP, 0);

    // u = sigmoid(G)*h + (1-sigmoid(G))*s_ ; uu = concat(u_fw, u_bw)
    k_combine<<<ROWS, 256, 0, stream>>>(Gf, Gb, hF, sfw, sbw, uu, uuB);

    // att_s = elu(uu @ Ws1 + b1) @ Ws + b
    k_gemm<D2P><<<gB, wv, 0, stream>>>(uuB, Ws1T, nullptr, Ws1bP, tF, tBf, D2P, 1);
    k_gemm<D2P><<<gB, wv, 0, stream>>>(tBf, WsT,  nullptr, WsbP,  atts, nullptr, D2P, 0);

    // s_s = sum_l uu * att_s ; y = relu(s_s @ F1 + b) @ F2 + b
    k_ss<<<BB, 512, 0, stream>>>(uu, atts, ssb);
    k_final<<<BB, 256, 0, stream>>>(ssb, F1w, F1b, F2w, F2b, y);
}
